// Encoder_trans_781684047976
// MI455X (gfx1250) — matrix operand order repaired and checked by reading, not compiled, not run on hardware
//
#include <hip/hip_runtime.h>
#include <cstddef>
#include <cstdint>

// ---------------------------------------------------------------------------
// Types for CDNA5 WMMA (wave32, v_wmma_f32_16x16x32_bf16)
// ---------------------------------------------------------------------------
typedef __bf16 bf16_t;
typedef __attribute__((ext_vector_type(16))) __bf16 v16bf;
typedef __attribute__((ext_vector_type(8)))  float  v8f;

union __attribute__((aligned(16))) BF8 {
    unsigned u[4];
    __bf16   h[8];
};

// float -> bf16, round-to-nearest-even, pure bit manipulation
__device__ __forceinline__ bf16_t f2b(float f) {
    unsigned x = __builtin_bit_cast(unsigned, f);
    unsigned r = (x + 0x7FFFu + ((x >> 16) & 1u)) >> 16;
    unsigned short hs = (unsigned short)r;
    return __builtin_bit_cast(bf16_t, hs);
}

// 16-element bf16 fragment from two 16-byte chunks (elems 0..7 = lo, 8..15 = hi)
__device__ __forceinline__ v16bf make_frag(const bf16_t* lo, const bf16_t* hi) {
    BF8 a = *(const BF8*)lo;
    BF8 b = *(const BF8*)hi;
    v16bf f;
#pragma unroll
    for (int i = 0; i < 8; i++) { f[i] = a.h[i]; f[i + 8] = b.h[i]; }
    return f;
}

__device__ __forceinline__ v8f zero_v8f() {
    v8f z;
#pragma unroll
    for (int i = 0; i < 8; i++) z[i] = 0.f;
    return z;
}

#define WMMA_BF16(A, Bf, C) \
    __builtin_amdgcn_wmma_f32_16x16x32_bf16(false, (A), false, (Bf), (short)0, (C), false, false)

// Problem constants
constexpr int Bc = 8, Sc = 1024, INc = 256, Hc = 1024, NHc = 16, HDc = 64, PFc = 4096, Lc = 6;
constexpr int NTOK = Bc * Sc; // 8192

// ---------------------------------------------------------------------------
// Tensor Data Mover (gfx1250): async 2D tile DMA global -> LDS, with in-flight
// LDS row padding (pad 4 DWORDs after every 16 DWORDs = 32 bf16 -> 40 bf16 rows).
// Toolchain uses the 6-arg form: (u32x4, i32x8, i32x4, i32x4, i32x8, i32 cpol).
// ---------------------------------------------------------------------------
#if defined(__has_builtin)
#if __has_builtin(__builtin_amdgcn_tensor_load_to_lds) && __has_builtin(__builtin_amdgcn_s_wait_tensorcnt)
#define HAVE_TDM 1
#endif
#endif

#ifdef HAVE_TDM
typedef unsigned int u32x4 __attribute__((ext_vector_type(4)));
typedef int i32x4 __attribute__((ext_vector_type(4)));
typedef int i32x8 __attribute__((ext_vector_type(8)));

// 2D tile load: tile_d0 x tile_d1 bf16 elements, row stride `stride0` elements
// in global memory, destination LDS rows padded 32 -> 40 elements.
__device__ __forceinline__ void tdm_load_tile_2d(const bf16_t* gsrc, unsigned lds_off,
                                                 unsigned tensor_d0, unsigned tensor_d1,
                                                 unsigned tile_d0, unsigned tile_d1,
                                                 unsigned stride0) {
    unsigned long long ga = (unsigned long long)(uintptr_t)gsrc;
    u32x4 g0;
    g0[0] = 1u;                                               // count=1, user desc
    g0[1] = lds_off;                                          // lds_addr (bytes)
    g0[2] = (unsigned)(ga & 0xFFFFFFFFu);                     // global_addr[31:0]
    g0[3] = (unsigned)((ga >> 32) & 0x1FFFFFFu) | (2u << 30); // ga[56:32] | type=2
    i32x8 g1;
    // data_size=1 (2B) | pad_enable | pad_interval=3 (16 DW) | pad_amount=3 (4 DW)
    g1[0] = (int)((1u << 16) | (1u << 20) | (3u << 22) | (3u << 25));
    g1[1] = (int)((tensor_d0 & 0xFFFFu) << 16);
    g1[2] = (int)(((tensor_d0 >> 16) & 0xFFFFu) | ((tensor_d1 & 0xFFFFu) << 16));
    g1[3] = (int)(((tensor_d1 >> 16) & 0xFFFFu) | ((tile_d0 & 0xFFFFu) << 16));
    g1[4] = (int)(tile_d1 & 0xFFFFu);                         // tile_dim2 = 0
    g1[5] = (int)stride0;                                     // dim0_stride[31:0]
    g1[6] = 0;
    g1[7] = 0;
    i32x4 z4;
    z4[0] = z4[1] = z4[2] = z4[3] = 0;
    i32x8 z8;
#pragma unroll
    for (int i = 0; i < 8; i++) z8[i] = 0;
    __builtin_amdgcn_tensor_load_to_lds(g0, g1, z4, z4, z8, 0);
}
#endif

// ---------------------------------------------------------------------------
// f32 -> bf16 conversion (grid-stride)
// ---------------------------------------------------------------------------
__global__ void cvt_f32_bf16(const float* __restrict__ in, bf16_t* __restrict__ out, size_t n) {
    size_t i = (size_t)blockIdx.x * blockDim.x + threadIdx.x;
    size_t stride = (size_t)gridDim.x * blockDim.x;
    for (; i < n; i += stride) out[i] = f2b(in[i]);
}

// ---------------------------------------------------------------------------
// Tiled bf16 GEMM with WMMA, double-buffered LDS (1 barrier per K-step).
// Block tile 128x128, K-step 32. 256 threads = 8 waves (4 row x 2 col); each
// wave owns a 32x64 sub-tile = 2x4 fragments of 16x16.
// A tile staged by TDM (if available) with in-flight padding; W tile staged
// manually transposed (sB[n][k]) for contiguous fragment reads.
// MODE 0: embedding  (acc+bias)*scale + pos -> f32 outf + bf16 outb
// MODE 1: QKV        acc+bias -> bf16 outb in [B,NH,S,HD]
// MODE 2: plain      acc+bias -> f32 outf
// MODE 3: relu       max(acc+bias,0) -> bf16 outb
// ---------------------------------------------------------------------------
template <int MODE>
__global__ void __launch_bounds__(256)
gemm_bf16(const bf16_t* __restrict__ A, const bf16_t* __restrict__ W,
          const float* __restrict__ bias, float* __restrict__ outf,
          bf16_t* __restrict__ outb, const float* __restrict__ pos,
          float scale, int M, int N, int K) {
    constexpr int BK = 32, PAD = 40; // 80B row stride: 16B aligned, bank-friendly
    __shared__ __attribute__((aligned(16))) bf16_t sA[2][128][PAD]; // sA[buf][m][k]
    __shared__ __attribute__((aligned(16))) bf16_t sB[2][128][PAD]; // sB[buf][n][k]

    const int tid = threadIdx.x;
    const int wave = tid >> 5, lane = tid & 31;
    const int g = lane >> 4, nl = lane & 15;
    const int wm = wave & 3, wn = wave >> 2;
    const int m0 = blockIdx.x * 128, n0 = blockIdx.y * 128;

    auto stage = [&](int buf, int k0) {
#ifdef HAVE_TDM
        if (wave == 0) {
            tdm_load_tile_2d(A + (size_t)m0 * K + k0,
                             (unsigned)(uintptr_t)(void*)&sA[buf][0][0],
                             (unsigned)K, (unsigned)M, 32u, 128u, (unsigned)K);
        }
#else
#pragma unroll
        for (int i = 0; i < 2; i++) {
            int seg = tid * 2 + i; // 0..511
            int row = seg >> 2, kk = (seg & 3) * 8;
            BF8 d = *(const BF8*)(A + (size_t)(m0 + row) * K + k0 + kk);
            *(BF8*)(&sA[buf][row][kk]) = d;
        }
#endif
        // W tile transposed: 32x128 -> sB[buf][n][k]
#pragma unroll
        for (int i = 0; i < 2; i++) {
            int seg = tid * 2 + i;
            int kk = seg >> 4, nn = (seg & 15) * 8;
            BF8 d = *(const BF8*)(W + (size_t)(k0 + kk) * N + n0 + nn);
#pragma unroll
            for (int j = 0; j < 8; j++) sB[buf][nn + j][kk] = d.h[j];
        }
    };

    v8f acc[2][4];
#pragma unroll
    for (int s = 0; s < 2; s++)
#pragma unroll
        for (int c = 0; c < 4; c++) acc[s][c] = zero_v8f();

    stage(0, 0);
#ifdef HAVE_TDM
    __builtin_amdgcn_s_wait_tensorcnt(0);
#endif
    __syncthreads();

    for (int k0 = 0; k0 < K; k0 += BK) {
        const int buf = (k0 >> 5) & 1;
        if (k0 + BK < K) stage(buf ^ 1, k0 + BK); // overlap next-tile staging

        v16bf af[2], bfr[4];
#pragma unroll
        for (int s = 0; s < 2; s++) {
            int m = wm * 32 + s * 16 + nl;
            af[s] = make_frag(&sA[buf][m][8 * g], &sA[buf][m][16 + 8 * g]);
        }
#pragma unroll
        for (int c = 0; c < 4; c++) {
            int n = wn * 64 + c * 16 + nl;
            bfr[c] = make_frag(&sB[buf][n][8 * g], &sB[buf][n][16 + 8 * g]);
        }
#pragma unroll
        for (int s = 0; s < 2; s++)
#pragma unroll
            for (int c = 0; c < 4; c++)
                acc[s][c] = WMMA_BF16(af[s], bfr[c], acc[s][c]);

#ifdef HAVE_TDM
        __builtin_amdgcn_s_wait_tensorcnt(0);
#endif
        __syncthreads();
    }

    // Epilogue. C layout: VGPR r holds rows r (lanes 0-15) / r+8 (lanes 16-31).
#pragma unroll
    for (int s = 0; s < 2; s++)
#pragma unroll
        for (int c = 0; c < 4; c++)
#pragma unroll
            for (int r = 0; r < 8; r++) {
                int row = m0 + wm * 32 + s * 16 + r + 8 * g;
                int col = n0 + wn * 64 + c * 16 + nl;
                float v = acc[s][c][r] + bias[col];
                if (MODE == 0) {
                    v = v * scale + pos[(size_t)(row & (Sc - 1)) * N + col];
                    outf[(size_t)row * N + col] = v;
                    outb[(size_t)row * N + col] = f2b(v);
                } else if (MODE == 1) {
                    int bi = row >> 10, sidx = row & (Sc - 1);
                    int hh = col >> 6, dd = col & (HDc - 1);
                    outb[(((size_t)(bi * NHc + hh) << 10) + sidx) * HDc + dd] = f2b(v);
                } else if (MODE == 2) {
                    outf[(size_t)row * N + col] = v;
                } else {
                    v = v > 0.f ? v : 0.f;
                    outb[(size_t)row * N + col] = f2b(v);
                }
            }
}

// ---------------------------------------------------------------------------
// Flash attention, one (b, h, 64-row q block) per 128-thread block (4 waves).
// K-tiles of 32 keys; V tile double-buffered in LDS (1 barrier per tile).
// P tile is wave-private LDS (same-wave DS ops are in-order: no barrier).
// scale = 1/sqrt(H) = 1/32 (the reference scales by sqrt(hidden), not head dim).
// ---------------------------------------------------------------------------
__global__ void __launch_bounds__(128)
flash_attn(const bf16_t* __restrict__ Q, const bf16_t* __restrict__ Kb,
           const bf16_t* __restrict__ V, bf16_t* __restrict__ O) {
    __shared__ __attribute__((aligned(16))) bf16_t vt[2][64][40];  // vt[buf][hd][key]
    __shared__ __attribute__((aligned(16))) bf16_t pp[4][16][40];  // per-wave P tile

    const int tid = threadIdx.x;
    const int wave = tid >> 5, lane = tid & 31;
    const int g = lane >> 4, nl = lane & 15;
    const int b = blockIdx.z, h = blockIdx.y;
    const int q0 = blockIdx.x * 64 + wave * 16;

    const size_t head = ((size_t)b * NHc + h) * Sc;
    const bf16_t* qbase = Q + head * HDc;
    const bf16_t* kbase = Kb + head * HDc;
    const bf16_t* vbase = V + head * HDc;

    // Cooperative V tile stage (32 keys x 64 hd), transposed into LDS
    auto stage_v = [&](int buf, int kt) {
        int key = tid & 31, hd0 = (tid >> 5) * 16;
        const bf16_t* p = vbase + (size_t)(kt + key) * HDc + hd0;
        BF8 d0 = *(const BF8*)p;
        BF8 d1 = *(const BF8*)(p + 8);
#pragma unroll
        for (int j = 0; j < 8; j++) {
            vt[buf][hd0 + j][key]     = d0.h[j];
            vt[buf][hd0 + 8 + j][key] = d1.h[j];
        }
    };

    // Preload Q A-fragments for hd halves [0,32) and [32,64)
    v16bf qf[2];
#pragma unroll
    for (int hh = 0; hh < 2; hh++) {
        const bf16_t* p = qbase + (size_t)(q0 + nl) * HDc + 32 * hh;
        qf[hh] = make_frag(p + 8 * g, p + 16 + 8 * g);
    }

    float mstat[8], lstat[8];
#pragma unroll
    for (int r = 0; r < 8; r++) { mstat[r] = -3.0e38f; lstat[r] = 0.f; }
    v8f oacc[4];
#pragma unroll
    for (int c = 0; c < 4; c++) oacc[c] = zero_v8f();

    stage_v(0, 0);
    __syncthreads();

    for (int kt = 0; kt < Sc; kt += 32) {
        const int buf = (kt >> 5) & 1;
        if (kt + 32 < Sc) stage_v(buf ^ 1, kt + 32); // overlap next V tile

        // Energy: two 16x16 fragments (keys kt..+16, kt+16..+32); K fragments
        // are contiguous 16B reads straight from global.
        v8f e[2];
        e[0] = zero_v8f(); e[1] = zero_v8f();
#pragma unroll
        for (int c = 0; c < 2; c++) {
#pragma unroll
            for (int hh = 0; hh < 2; hh++) {
                const bf16_t* p = kbase + (size_t)(kt + c * 16 + nl) * HDc + 32 * hh;
                v16bf kf = make_frag(p + 8 * g, p + 16 + 8 * g);
                e[c] = WMMA_BF16(qf[hh], kf, e[c]);
            }
        }

        // Online softmax (row stats replicated within each 16-lane half)
        float corr[8];
#pragma unroll
        for (int r = 0; r < 8; r++) {
            float t0 = e[0][r] * 0.03125f;
            float t1 = e[1][r] * 0.03125f;
            float tmax = fmaxf(t0, t1);
#pragma unroll
            for (int off = 8; off > 0; off >>= 1) tmax = fmaxf(tmax, __shfl_xor(tmax, off, 16));
            float nm = fmaxf(mstat[r], tmax);
            float p0 = __expf(t0 - nm);
            float p1 = __expf(t1 - nm);
            float rs = p0 + p1;
#pragma unroll
            for (int off = 8; off > 0; off >>= 1) rs += __shfl_xor(rs, off, 16);
            corr[r] = __expf(mstat[r] - nm);
            lstat[r] = lstat[r] * corr[r] + rs;
            mstat[r] = nm;
            int m = r + 8 * g;
            pp[wave][m][nl]      = f2b(p0);
            pp[wave][m][16 + nl] = f2b(p1);
        }

        // Rescale O accumulators, then O += P @ V (V from current buffer)
#pragma unroll
        for (int c = 0; c < 4; c++)
#pragma unroll
            for (int r = 0; r < 8; r++) oacc[c][r] *= corr[r];

        v16bf pf = make_frag(&pp[wave][nl][8 * g], &pp[wave][nl][16 + 8 * g]);
#pragma unroll
        for (int c = 0; c < 4; c++) {
            v16bf vf = make_frag(&vt[buf][c * 16 + nl][8 * g], &vt[buf][c * 16 + nl][16 + 8 * g]);
            oacc[c] = WMMA_BF16(pf, vf, oacc[c]);
        }
        __syncthreads(); // next V buffer complete; this buffer free for reuse
    }

    // Normalize and write attention output in [B,S,H] bf16 (head-concat layout)
#pragma unroll
    for (int c = 0; c < 4; c++)
#pragma unroll
        for (int r = 0; r < 8; r++) {
            int m = r + 8 * g;
            int qrow = q0 + m;
            float v = oacc[c][r] / lstat[r];
            O[((size_t)b * Sc + qrow) * Hc + h * HDc + c * 16 + nl] = f2b(v);
        }
}

// ---------------------------------------------------------------------------
// Fused residual add + LayerNorm (f32), writes f32 (in place) + bf16 mirror.
// ---------------------------------------------------------------------------
__global__ void __launch_bounds__(256)
add_ln(float* __restrict__ x, const float* __restrict__ d,
       const float* __restrict__ gw, const float* __restrict__ bw,
       bf16_t* __restrict__ xb) {
    __shared__ float red[8];
    const int row = blockIdx.x, tid = threadIdx.x;
    const int wave = tid >> 5, lane = tid & 31;
    const size_t base = (size_t)row * Hc;

    float v[4];
    float s = 0.f;
#pragma unroll
    for (int i = 0; i < 4; i++) {
        int c = tid + 256 * i;
        v[i] = x[base + c] + d[base + c];
        s += v[i];
    }
#pragma unroll
    for (int off = 16; off > 0; off >>= 1) s += __shfl_xor(s, off);
    if (lane == 0) red[wave] = s;
    __syncthreads();
    float tot = 0.f;
#pragma unroll
    for (int i = 0; i < 8; i++) tot += red[i];
    float mean = tot * (1.f / (float)Hc);

    float sq = 0.f;
#pragma unroll
    for (int i = 0; i < 4; i++) { float t = v[i] - mean; sq += t * t; }
#pragma unroll
    for (int off = 16; off > 0; off >>= 1) sq += __shfl_xor(sq, off);
    __syncthreads();
    if (lane == 0) red[wave] = sq;
    __syncthreads();
    float tot2 = 0.f;
#pragma unroll
    for (int i = 0; i < 8; i++) tot2 += red[i];
    float inv = rsqrtf(tot2 * (1.f / (float)Hc) + 1e-5f);

#pragma unroll
    for (int i = 0; i < 4; i++) {
        int c = tid + 256 * i;
        float y = (v[i] - mean) * inv * gw[c] + bw[c];
        x[base + c] = y;
        xb[base + c] = f2b(y);
    }
}

// ---------------------------------------------------------------------------
// Host orchestration
// ---------------------------------------------------------------------------
extern "C" void kernel_launch(void* const* d_in, const int* in_sizes, int n_in,
                              void* d_out, int out_size, void* d_ws, size_t ws_size,
                              hipStream_t stream) {
    (void)in_sizes; (void)n_in; (void)out_size; (void)ws_size;
    const float* input_x = (const float*)d_in[0];
    const float* emb_W   = (const float*)d_in[1];
    const float* emb_b   = (const float*)d_in[2];
    const float* pos_tab = (const float*)d_in[3];
    const float* Wq = (const float*)d_in[4];
    const float* bq = (const float*)d_in[5];
    const float* Wk = (const float*)d_in[6];
    const float* bk = (const float*)d_in[7];
    const float* Wv = (const float*)d_in[8];
    const float* bv = (const float*)d_in[9];
    const float* Wo = (const float*)d_in[10];
    const float* bo = (const float*)d_in[11];
    const float* ln1_g = (const float*)d_in[12];
    const float* ln1_b = (const float*)d_in[13];
    const float* W1 = (const float*)d_in[14];
    const float* b1 = (const float*)d_in[15];
    const float* W2 = (const float*)d_in[16];
    const float* b2 = (const float*)d_in[17];
    const float* ln2_g = (const float*)d_in[18];
    const float* ln2_b = (const float*)d_in[19];

    float* x = (float*)d_out; // running f32 activation lives in d_out

    char* ws = (char*)d_ws;
    size_t off = 0;
    auto alloc = [&](size_t bytes) -> char* {
        char* p = ws + off;
        off += (bytes + 255) & ~(size_t)255;
        return p;
    };

    bf16_t* xb    = (bf16_t*)alloc((size_t)NTOK * Hc * 2);
    bf16_t* qb    = (bf16_t*)alloc((size_t)NTOK * Hc * 2);
    bf16_t* kb    = (bf16_t*)alloc((size_t)NTOK * Hc * 2);
    bf16_t* vb    = (bf16_t*)alloc((size_t)NTOK * Hc * 2);
    bf16_t* ab    = (bf16_t*)alloc((size_t)NTOK * Hc * 2);
    float*  obuf  = (float*) alloc((size_t)NTOK * Hc * 4);
    bf16_t* f1b   = (bf16_t*)alloc((size_t)NTOK * PFc * 2);
    bf16_t* inxb  = (bf16_t*)alloc((size_t)NTOK * INc * 2);
    bf16_t* embWb = (bf16_t*)alloc((size_t)INc * Hc * 2);
    bf16_t* Wqb   = (bf16_t*)alloc((size_t)Lc * Hc * Hc * 2);
    bf16_t* Wkb   = (bf16_t*)alloc((size_t)Lc * Hc * Hc * 2);
    bf16_t* Wvb   = (bf16_t*)alloc((size_t)Lc * Hc * Hc * 2);
    bf16_t* Wob   = (bf16_t*)alloc((size_t)Lc * Hc * Hc * 2);
    bf16_t* W1b   = (bf16_t*)alloc((size_t)Lc * Hc * PFc * 2);
    bf16_t* W2b   = (bf16_t*)alloc((size_t)Lc * PFc * Hc * 2);

    auto cvt = [&](const float* src, bf16_t* dst, size_t n) {
        size_t blocks = (n + 255) / 256;
        if (blocks > 16384) blocks = 16384;
        cvt_f32_bf16<<<dim3((unsigned)blocks), dim3(256), 0, stream>>>(src, dst, n);
    };
    cvt(input_x, inxb, (size_t)NTOK * INc);
    cvt(emb_W, embWb, (size_t)INc * Hc);
    cvt(Wq, Wqb, (size_t)Lc * Hc * Hc);
    cvt(Wk, Wkb, (size_t)Lc * Hc * Hc);
    cvt(Wv, Wvb, (size_t)Lc * Hc * Hc);
    cvt(Wo, Wob, (size_t)Lc * Hc * Hc);
    cvt(W1, W1b, (size_t)Lc * Hc * PFc);
    cvt(W2, W2b, (size_t)Lc * PFc * Hc);

    const dim3 blk256(256), blk128(128);
    const dim3 gH(NTOK / 128, Hc / 128);   // (64, 8)
    const dim3 gP(NTOK / 128, PFc / 128);  // (64, 32)
    const float sqrtH = 32.0f;

    // Embedding: x = (input @ embW + b) * sqrt(H) + pos
    gemm_bf16<0><<<gH, blk256, 0, stream>>>(inxb, embWb, emb_b, x, xb, pos_tab,
                                            sqrtH, NTOK, Hc, INc);

    const size_t WH = (size_t)Hc * Hc;
    const size_t WP = (size_t)Hc * PFc;
    for (int l = 0; l < Lc; l++) {
        gemm_bf16<1><<<gH, blk256, 0, stream>>>(xb, Wqb + l * WH, bq + l * Hc,
                                                nullptr, qb, nullptr, 0.f, NTOK, Hc, Hc);
        gemm_bf16<1><<<gH, blk256, 0, stream>>>(xb, Wkb + l * WH, bk + l * Hc,
                                                nullptr, kb, nullptr, 0.f, NTOK, Hc, Hc);
        gemm_bf16<1><<<gH, blk256, 0, stream>>>(xb, Wvb + l * WH, bv + l * Hc,
                                                nullptr, vb, nullptr, 0.f, NTOK, Hc, Hc);

        flash_attn<<<dim3(Sc / 64, NHc, Bc), blk128, 0, stream>>>(qb, kb, vb, ab);

        gemm_bf16<2><<<gH, blk256, 0, stream>>>(ab, Wob + l * WH, bo + l * Hc,
                                                obuf, nullptr, nullptr, 0.f, NTOK, Hc, Hc);
        add_ln<<<dim3(NTOK), blk256, 0, stream>>>(x, obuf, ln1_g + l * Hc, ln1_b + l * Hc, xb);

        gemm_bf16<3><<<gP, blk256, 0, stream>>>(xb, W1b + l * WP, b1 + l * PFc,
                                                nullptr, f1b, nullptr, 0.f, NTOK, PFc, Hc);
        gemm_bf16<2><<<gH, blk256, 0, stream>>>(f1b, W2b + l * WP, b2 + l * Hc,
                                                obuf, nullptr, nullptr, 0.f, NTOK, Hc, PFc);
        add_ln<<<dim3(NTOK), blk256, 0, stream>>>(x, obuf, ln2_g + l * Hc, ln2_b + l * Hc, xb);
    }
}
